// WindowAttentionWithRPB_10316511445030
// MI455X (gfx1250) — compile-verified, matrix operand order checked by reading, not run on hardware
//
#include <hip/hip_runtime.h>

typedef __bf16 v16bf __attribute__((ext_vector_type(16)));
typedef float  v8f   __attribute__((ext_vector_type(8)));

#define DIMC  256
#define NH    8
#define HD    32
#define NWIN  144   // tokens per window
#define NWP   160   // padded token count for P@V K-loop (5 * 32)
#define NWINBLK 2048

// ---- LDS layout (bytes) ----
#define XS_OFF   0                            // bf16 [144][256]  x window
#define QS_OFF   (XS_OFF + NWIN*DIMC*2)       // bf16 [144][32]   q (scaled)
#define KS_OFF   (QS_OFF + NWIN*HD*2)         // bf16 [144][32]   k (token-major)
#define VT_OFF   (KS_OFF + NWIN*HD*2)         // bf16 [32][160]   v transposed (dim-major)
#define SF_OFF   (VT_OFF + HD*NWP*2)          // f32  [144][160]  scores
#define PB_OFF   (SF_OFF + NWIN*NWP*4)        // bf16 [144][160]  probs
#define OB_OFF   (PB_OFF + NWIN*NWP*2)        // bf16 [144][256]  attn out
#define LDS_BYTES (OB_OFF + NWIN*DIMC*2)      // = 314368
// Phase-A f32 staging buffer reuses the (then-dead) Sf/Pb region:
// needs 144*256*4 = 147456 B; SF_OFF+147456 = 249856 <= LDS_BYTES.  OK.

// ---- workspace layout (bytes) ----
#define WS_WQKVT 0                            // bf16 [768][256]
#define WS_WPROJT (WS_WQKVT + 768*256*2)      // bf16 [256][256]
#define WS_BIAS   (WS_WPROJT + 256*256*2)     // f32  [8][144][144]

__device__ __forceinline__ __bf16 f2bf(float f) {
    unsigned int u = __float_as_uint(f);
    unsigned int r = (u + 0x7FFFu + ((u >> 16) & 1u)) >> 16;
    union { unsigned short s; __bf16 b; } cv;
    cv.s = (unsigned short)r;
    return cv.b;
}

// A-fragment (16x32 bf16) from a row-major [rows][ld] buffer.
// Lane l<16 holds row (row0+l), K chunks {k0..k0+7, k0+16..k0+23};
// lane l>=16 holds row (row0+l-16), K chunks {k0+8.., k0+24..}. (ISA 7.12.2)
// Same pattern loads a B-fragment when the B matrix is stored N-major.
__device__ __forceinline__ v16bf load_frag(const __bf16* base, int ld,
                                           int row0, int k0, int lane) {
    int half = lane >> 4;
    int r    = lane & 15;
    const __bf16* p = base + (size_t)(row0 + r) * ld + k0 + half * 8;
    union { uint4 q[2]; v16bf v; } u;
    u.q[0] = *(const uint4*)(p);
    u.q[1] = *(const uint4*)(p + 16);
    return u.v;
}

__device__ __forceinline__ v8f wmma_bf16(v16bf a, v16bf b, v8f c) {
    return __builtin_amdgcn_wmma_f32_16x16x32_bf16(
        false, a, false, b, (short)0, c, false, false);
}

// ---------------- prep kernels ----------------
__global__ __launch_bounds__(256)
void prep_weights_kernel(const float* __restrict__ qkv_w,   // [256][768]
                         const float* __restrict__ proj_w,  // [256][256]
                         __bf16* __restrict__ wqkvT,        // [768][256]
                         __bf16* __restrict__ wprojT) {     // [256][256]
    int idx = blockIdx.x * 256 + threadIdx.x;
    if (idx < 768 * 256) {
        int o = idx / 256, i = idx % 256;
        wqkvT[o * 256 + i] = f2bf(qkv_w[i * 768 + o]);
    }
    if (idx < 256 * 256) {
        int o = idx / 256, i = idx % 256;
        wprojT[o * 256 + i] = f2bf(proj_w[i * 256 + o]);
    }
}

__global__ __launch_bounds__(256)
void prep_bias_kernel(const float* __restrict__ rpb,  // [(2W-1)^2][8]
                      const int* __restrict__ rel,    // [144][144]
                      float* __restrict__ biasbuf) {  // [8][144][144]
    int idx = blockIdx.x * 256 + threadIdx.x;
    if (idx < NH * NWIN * NWIN) {
        int h  = idx / (NWIN * NWIN);
        int ij = idx % (NWIN * NWIN);
        biasbuf[idx] = rpb[rel[ij] * NH + h];
    }
}

// ---------------- fused per-window attention ----------------
__global__ __launch_bounds__(256)
void fused_window_attn_kernel(const float* __restrict__ x,       // [2048][144][256]
                              const float* __restrict__ qkv_b,   // [768]
                              const float* __restrict__ proj_b,  // [256]
                              const __bf16* __restrict__ wqkvT,  // [768][256]
                              const __bf16* __restrict__ wprojT, // [256][256]
                              const float* __restrict__ biasbuf, // [8][144][144]
                              float* __restrict__ out) {         // [2048][144][256]
    extern __shared__ char smem[];
    __bf16* xs = (__bf16*)(smem + XS_OFF);
    __bf16* qs = (__bf16*)(smem + QS_OFF);
    __bf16* ks = (__bf16*)(smem + KS_OFF);
    __bf16* vT = (__bf16*)(smem + VT_OFF);
    float*  Sf = (float*) (smem + SF_OFF);
    __bf16* Pb = (__bf16*)(smem + PB_OFF);
    __bf16* ob = (__bf16*)(smem + OB_OFF);
    float*  xstage = (float*)(smem + SF_OFF);  // phase-A only; Sf/Pb dead then

    const int win  = blockIdx.x;
    const int tid  = threadIdx.x;
    const int lane = tid & 31;
    // readfirstlane => SGPR wave id => scalar branches, full-EXEC WMMA
    const int wave = __builtin_amdgcn_readfirstlane(tid >> 5);
    const int half = lane >> 4;
    const int col  = lane & 15;

    // warm L2/WGP$ with the transposed weights while phase A streams x
    __builtin_prefetch(((const char*)wqkvT) + tid * 1536, 0, 3);
    __builtin_prefetch(((const char*)wprojT) + tid * 512, 0, 3);

    // ---- Phase A: async-stage raw f32 x window into LDS (ASYNCcnt path) ----
    const float* xw = x + (size_t)win * NWIN * DIMC;
    {
        unsigned lds_base = (unsigned)(uintptr_t)xstage;
        #pragma unroll
        for (int it = 0; it < (NWIN * DIMC * 4) / (256 * 16); ++it) {  // 36
            int byteoff = (it * 256 + tid) * 16;
            unsigned lds_addr = lds_base + byteoff;
            const char* gaddr = (const char*)xw + byteoff;
            asm volatile("global_load_async_to_lds_b128 %0, %1, off"
                         :: "v"(lds_addr), "v"(gaddr)
                         : "memory");
        }
        asm volatile("s_wait_asynccnt 0x0" ::: "memory");
    }
    __syncthreads();

    // convert staged f32 -> bf16 xs (LDS -> LDS)
    #pragma unroll
    for (int it = 0; it < (NWIN * DIMC / 4) / 256; ++it) {  // 36
        int e4 = tid + it * 256;
        float4 v = ((const float4*)xstage)[e4];
        __bf16* d = xs + e4 * 4;
        d[0] = f2bf(v.x); d[1] = f2bf(v.y); d[2] = f2bf(v.z); d[3] = f2bf(v.w);
    }
    __syncthreads();

    const float scale = 0.17677669529663687f;  // 32^-0.5

    for (int h = 0; h < NH; ++h) {
        // zero the vT token-pad region [144,160) so padded K contributes 0
        for (int e = tid; e < HD * (NWP - NWIN); e += 256) {
            int d = e / (NWP - NWIN), t = e % (NWP - NWIN);
            vT[d * NWP + NWIN + t] = f2bf(0.0f);
        }
        __syncthreads();

        // ---- QKV projection for this head: 3 mats x 9 Mtiles x 2 Ntiles ----
        for (int j = wave; j < 54; j += 8) {
            int mat = j / 18, rem = j % 18;
            int m0 = (rem >> 1) * 16;
            int ocol = mat * DIMC + h * HD + (rem & 1) * 16;  // row base in wqkvT
            v8f acc = {};
            #pragma unroll
            for (int kc = 0; kc < 8; ++kc) {
                v16bf a = load_frag(xs, DIMC, m0, kc * 32, lane);
                v16bf b = load_frag(wqkvT, DIMC, ocol, kc * 32, lane);
                acc = wmma_bf16(a, b, acc);
            }
            float bia = qkv_b[ocol + col];
            int nn = (rem & 1) * 16 + col;
            if (mat == 0) {
                #pragma unroll
                for (int i = 0; i < 8; ++i)
                    qs[(m0 + i + 8 * half) * HD + nn] = f2bf((acc[i] + bia) * scale);
            } else if (mat == 1) {
                #pragma unroll
                for (int i = 0; i < 8; ++i)
                    ks[(m0 + i + 8 * half) * HD + nn] = f2bf(acc[i] + bia);
            } else {
                #pragma unroll
                for (int i = 0; i < 8; ++i)  // transposed store: vT[dim][token]
                    vT[nn * NWP + (m0 + i + 8 * half)] = f2bf(acc[i] + bia);
            }
        }
        __syncthreads();

        // ---- S = q @ k^T : 9x9 tiles, K = HD = 32 (single WMMA each) ----
        for (int j = wave; j < 81; j += 8) {
            int m0 = (j / 9) * 16, n0 = (j % 9) * 16;
            v16bf a = load_frag(qs, HD, m0, 0, lane);
            v16bf b = load_frag(ks, HD, n0, 0, lane);  // B from token-major k == k^T
            v8f acc = {};
            acc = wmma_bf16(a, b, acc);
            #pragma unroll
            for (int i = 0; i < 8; ++i)
                Sf[(m0 + i + 8 * half) * NWP + n0 + col] = acc[i];
        }
        __syncthreads();

        // ---- bias + softmax, one query row per thread ----
        if (tid < NWIN) {
            const float* brow = biasbuf + ((size_t)h * NWIN + tid) * NWIN;
            float* srow = Sf + tid * NWP;
            float m = -3.4e38f;
            for (int jj = 0; jj < NWIN; ++jj) {
                float s = srow[jj] + brow[jj];
                srow[jj] = s;
                m = fmaxf(m, s);
            }
            float sum = 0.0f;
            for (int jj = 0; jj < NWIN; ++jj) {
                float e = __expf(srow[jj] - m);
                srow[jj] = e;
                sum += e;
            }
            float inv = 1.0f / sum;
            __bf16* prow = Pb + tid * NWP;
            for (int jj = 0; jj < NWIN; ++jj) prow[jj] = f2bf(srow[jj] * inv);
            for (int jj = NWIN; jj < NWP; ++jj) prow[jj] = f2bf(0.0f);
        }
        __syncthreads();

        // ---- O_h = P @ V : 9 Mtiles x 2 Ntiles, K = 160 (5 WMMA steps) ----
        for (int j = wave; j < 18; j += 8) {
            int m0 = (j >> 1) * 16, n0 = (j & 1) * 16;
            v8f acc = {};
            #pragma unroll
            for (int kc = 0; kc < 5; ++kc) {
                v16bf a = load_frag(Pb, NWP, m0, kc * 32, lane);
                v16bf b = load_frag(vT, NWP, n0, kc * 32, lane);  // dim-major V == V^T
                acc = wmma_bf16(a, b, acc);
            }
            #pragma unroll
            for (int i = 0; i < 8; ++i)
                ob[(m0 + i + 8 * half) * DIMC + h * HD + n0 + col] = f2bf(acc[i]);
        }
        __syncthreads();  // before next head reuses qs/ks/vT/Sf/Pb
    }

    // ---- final projection: ob(144x256) @ proj_w + b -> out ----
    float* ow = out + (size_t)win * NWIN * DIMC;
    for (int j = wave; j < 144; j += 8) {
        int m0 = (j / 16) * 16, n0 = (j % 16) * 16;
        v8f acc = {};
        #pragma unroll
        for (int kc = 0; kc < 8; ++kc) {
            v16bf a = load_frag(ob, DIMC, m0, kc * 32, lane);
            v16bf b = load_frag(wprojT, DIMC, n0, kc * 32, lane);
            acc = wmma_bf16(a, b, acc);
        }
        float bia = proj_b[n0 + col];
        #pragma unroll
        for (int i = 0; i < 8; ++i)
            ow[(size_t)(m0 + i + 8 * half) * DIMC + n0 + col] = acc[i] + bia;
    }
}

extern "C" void kernel_launch(void* const* d_in, const int* in_sizes, int n_in,
                              void* d_out, int out_size, void* d_ws, size_t ws_size,
                              hipStream_t stream) {
    const float* x      = (const float*)d_in[0];
    const float* qkv_w  = (const float*)d_in[1];
    const float* qkv_b  = (const float*)d_in[2];
    const float* proj_w = (const float*)d_in[3];
    const float* proj_b = (const float*)d_in[4];
    const float* rpb    = (const float*)d_in[5];
    const int*   rel    = (const int*)d_in[6];

    char* ws = (char*)d_ws;
    __bf16* wqkvT  = (__bf16*)(ws + WS_WQKVT);
    __bf16* wprojT = (__bf16*)(ws + WS_WPROJT);
    float*  biasbuf = (float*)(ws + WS_BIAS);

    prep_weights_kernel<<<768, 256, 0, stream>>>(qkv_w, proj_w, wqkvT, wprojT);
    prep_bias_kernel<<<(NH * NWIN * NWIN + 255) / 256, 256, 0, stream>>>(rpb, rel, biasbuf);

    fused_window_attn_kernel<<<NWINBLK, 256, LDS_BYTES, stream>>>(
        x, qkv_b, proj_b, wqkvT, wprojT, biasbuf, (float*)d_out);
}
